// FNO_RC_1D_51376398795184
// MI455X (gfx1250) — matrix operand order repaired and checked by reading, not compiled, not run on hardware
//
#include <hip/hip_runtime.h>
#include <hip/hip_bf16.h>

typedef __attribute__((ext_vector_type(16))) _Float16 v16h;
typedef __attribute__((ext_vector_type(8)))  _Float16 v8h;
typedef __attribute__((ext_vector_type(4)))  _Float16 v4h;
typedef __attribute__((ext_vector_type(8)))  float    v8f;

#define BB   32
#define SS   8192
#define WID  128
#define NMODES 32
#define PITCH 40   // LDS tile row pitch (halves): 80B, multiple of 16B

__device__ __forceinline__ float gelu_f(float x) {
  return 0.5f * x * (1.0f + erff(x * 0.70710678118654752f));
}

// ---- WMMA fragment loaders (wave32, v_wmma_f32_16x16x32_f16) -------------
// A 16x32 (MxK), LDS row-major [16][PITCH]:
//   elems 0..7 = row[kh*8 .. +7], elems 8..15 = row[16+kh*8 .. +7]
__device__ __forceinline__ v16h frag_A(const _Float16* As) {
  const int lane = threadIdx.x & 31;
  const _Float16* row = As + (lane & 15) * PITCH + ((lane >> 4) * 8);
  v8h lo = *(const v8h*)(row);
  v8h hi = *(const v8h*)(row + 16);
  return __builtin_shufflevector(lo, hi, 0, 1, 2, 3, 4, 5, 6, 7,
                                 8, 9, 10, 11, 12, 13, 14, 15);
}
// B 32x16 (KxN) from transposed LDS tile Bt[n][PITCH]:
//   lane n = lane&15, elems 0..7 -> K = (lane>>4)*8 + e, elems 8..15 -> K = 16 + (lane>>4)*8 + (e-8)
__device__ __forceinline__ v16h frag_B(const _Float16* Bt) {
  const int lane = threadIdx.x & 31;
  const _Float16* row = Bt + (lane & 15) * PITCH + ((lane >> 4) * 8);
  v8h lo = *(const v8h*)(row);
  v8h hi = *(const v8h*)(row + 16);
  return __builtin_shufflevector(lo, hi, 0, 1, 2, 3, 4, 5, 6, 7,
                                 8, 9, 10, 11, 12, 13, 14, 15);
}

// ---- generic WMMA GEMM: C = act(A[M,K]*B[K,N] + biasN) -------------------
// 256 threads = 8 waves in 2(M) x 4(N) grid; each wave owns (BM/2)x(BN/4).
// Big config <64,128>: 4 wmma/wave/K-step. Small config <32,64>: 1.
// ALL dims exact multiples: M%BM==0 (grid.y), N%BN==0 (grid.x), K%32==0.
template <int BM, int BN>
__global__ __launch_bounds__(256) void gemm_wmma(
    const _Float16* __restrict__ A, int lda, long long sA,
    const _Float16* __restrict__ B, int ldb, long long sB,
    float* Cf, _Float16* Ch, int ldc, long long sC,
    const float* __restrict__ biasN, int K, int act) {
  constexpr int TM = BM / 2, TN = BN / 4;
  constexpr int NM = TM / 16, NN = TN / 16;
  constexpr int AH = BM / 8;   // A halves per thread (BM*32/256)
  constexpr int BH = BN / 8;   // B halves per thread (32*BN/256)
  __shared__ __align__(16) _Float16 As[BM * PITCH];
  __shared__ __align__(16) _Float16 Bt[BN * PITCH];
  const int tid = threadIdx.x;
  const int wave = tid >> 5;
  const int wm = (wave >> 2) * TM;
  const int wn = (wave & 3) * TN;
  const int m0 = blockIdx.y * BM;
  const int n0 = blockIdx.x * BN;
  const long long z = blockIdx.z;
  A += z * sA; B += z * sB;
  const int ar = (tid * AH) / 32, ac = (tid * AH) % 32;
  const int br = (tid * BH) / BN, bc = (tid * BH) % BN;
  v8f acc[NM][NN] = {};
  for (int k0 = 0; k0 < K; k0 += 32) {
    const _Float16* Ap = A + (long long)(m0 + ar) * lda + k0 + ac;
    if constexpr (AH == 8) *(v8h*)(As + ar * PITCH + ac) = *(const v8h*)(Ap);
    else                   *(v4h*)(As + ar * PITCH + ac) = *(const v4h*)(Ap);
    const _Float16* Bp = B + (long long)(k0 + br) * ldb + n0 + bc;
    {
      v8h b0 = *(const v8h*)(Bp);
#pragma unroll
      for (int j = 0; j < 8; ++j) Bt[(bc + j) * PITCH + br] = b0[j];
      if constexpr (BH == 16) {
        v8h b1 = *(const v8h*)(Bp + 8);
#pragma unroll
        for (int j = 0; j < 8; ++j) Bt[(bc + 8 + j) * PITCH + br] = b1[j];
      }
    }
    if (k0 + 32 < K) {
      __builtin_prefetch(Ap + 32, 0, 1);
      __builtin_prefetch(Bp + 32LL * ldb, 0, 1);
    }
    __syncthreads();
    v16h af[NM], bf[NN];
#pragma unroll
    for (int i = 0; i < NM; ++i) af[i] = frag_A(As + (wm + 16 * i) * PITCH);
#pragma unroll
    for (int j = 0; j < NN; ++j) bf[j] = frag_B(Bt + (wn + 16 * j) * PITCH);
#pragma unroll
    for (int i = 0; i < NM; ++i)
#pragma unroll
      for (int j = 0; j < NN; ++j)
        acc[i][j] = __builtin_amdgcn_wmma_f32_16x16x32_f16(
            false, af[i], false, bf[j], (short)0, acc[i][j], false, false);
    __syncthreads();
  }
  const int lane = tid & 31;
  const int n = lane & 15, kh = lane >> 4;
#pragma unroll
  for (int i = 0; i < NM; ++i)
#pragma unroll
    for (int j = 0; j < NN; ++j) {
      const int gc = n0 + wn + 16 * j + n;
#pragma unroll
      for (int vi = 0; vi < 8; ++vi) {
        int gm = m0 + wm + 16 * i + vi + 8 * kh;
        float x = acc[i][j][vi];
        if (biasN) x += biasN[gc];
        if (act) x = gelu_f(x);
        long long off = z * sC + (long long)gm * ldc + gc;
        if (Cf) Cf[off] = x;
        if (Ch) Ch[off] = (_Float16)x;
      }
    }
}

// ---- fused FNO block: h_out = act(invDFT(Xm) + W*h_in + bias) ------------
// Block tile 64(c) x 128(s); 8 waves 2x4, each 32x32 (2x2 wmma).
// grid: (S/128, C/64, B). Two K loops share the accumulators.
__global__ __launch_bounds__(256) void fno_block(
    const _Float16* __restrict__ hin,   // [B][128][S]
    const _Float16* __restrict__ Wl,    // [128o][128i]
    const _Float16* __restrict__ Xm,    // [B*128][64] (scaled modes)
    const _Float16* __restrict__ Binv,  // [64][S]
    const float* __restrict__ bias,     // [128]
    _Float16* __restrict__ hout, int do_gelu) {
  __shared__ __align__(16) _Float16 As[64 * PITCH];
  __shared__ __align__(16) _Float16 Bt[128 * PITCH];
  const int tid = threadIdx.x;
  const int wave = tid >> 5;
  const int wm = (wave >> 2) * 32;
  const int wn = (wave & 3) * 32;
  const int s0 = blockIdx.x * 128;
  const int c0 = blockIdx.y * 64;
  const long long b = blockIdx.z;
  const int ar = (tid * 8) >> 5, ac = (tid * 8) & 31;      // A: 64x32 tile
  const int br = (tid * 16) >> 7, bc = (tid * 16) & 127;   // B: 32x128 tile
  v8f acc[2][2] = {};
  // ---- pass 1: pointwise conv, K = 128 input channels --------------------
  for (int k0 = 0; k0 < 128; k0 += 32) {
    *(v8h*)(As + ar * PITCH + ac) =
        *(const v8h*)(Wl + (c0 + ar) * 128 + k0 + ac);
    const _Float16* Bp = hin + (b * WID + k0 + br) * SS + s0 + bc;
    v8h b0 = *(const v8h*)(Bp);
    v8h b1 = *(const v8h*)(Bp + 8);
#pragma unroll
    for (int j = 0; j < 8; ++j) Bt[(bc + j) * PITCH + br] = b0[j];
#pragma unroll
    for (int j = 0; j < 8; ++j) Bt[(bc + 8 + j) * PITCH + br] = b1[j];
    __syncthreads();
    v16h af[2], bf[2];
#pragma unroll
    for (int i = 0; i < 2; ++i) af[i] = frag_A(As + (wm + 16 * i) * PITCH);
#pragma unroll
    for (int j = 0; j < 2; ++j) bf[j] = frag_B(Bt + (wn + 16 * j) * PITCH);
#pragma unroll
    for (int i = 0; i < 2; ++i)
#pragma unroll
      for (int j = 0; j < 2; ++j)
        acc[i][j] = __builtin_amdgcn_wmma_f32_16x16x32_f16(
            false, af[i], false, bf[j], (short)0, acc[i][j], false, false);
    __syncthreads();
  }
  // ---- pass 2: inverse truncated DFT, K = 64 (32 modes re/im) ------------
  for (int k0 = 0; k0 < 64; k0 += 32) {
    *(v8h*)(As + ar * PITCH + ac) =
        *(const v8h*)(Xm + (b * WID + c0 + ar) * 64 + k0 + ac);
    const _Float16* Bp = Binv + (long long)(k0 + br) * SS + s0 + bc;
    v8h b0 = *(const v8h*)(Bp);
    v8h b1 = *(const v8h*)(Bp + 8);
#pragma unroll
    for (int j = 0; j < 8; ++j) Bt[(bc + j) * PITCH + br] = b0[j];
#pragma unroll
    for (int j = 0; j < 8; ++j) Bt[(bc + 8 + j) * PITCH + br] = b1[j];
    __syncthreads();
    v16h af[2], bf[2];
#pragma unroll
    for (int i = 0; i < 2; ++i) af[i] = frag_A(As + (wm + 16 * i) * PITCH);
#pragma unroll
    for (int j = 0; j < 2; ++j) bf[j] = frag_B(Bt + (wn + 16 * j) * PITCH);
#pragma unroll
    for (int i = 0; i < 2; ++i)
#pragma unroll
      for (int j = 0; j < 2; ++j)
        acc[i][j] = __builtin_amdgcn_wmma_f32_16x16x32_f16(
            false, af[i], false, bf[j], (short)0, acc[i][j], false, false);
    __syncthreads();
  }
  const int lane = tid & 31;
  const int n = lane & 15, kh = lane >> 4;
#pragma unroll
  for (int i = 0; i < 2; ++i)
#pragma unroll
    for (int j = 0; j < 2; ++j)
#pragma unroll
      for (int vi = 0; vi < 8; ++vi) {
        int cc = c0 + wm + 16 * i + vi + 8 * kh;
        float x = acc[i][j][vi] + bias[cc];
        if (do_gelu) x = gelu_f(x);
        hout[(b * WID + cc) * SS + s0 + wn + 16 * j + n] = (_Float16)x;
      }
}

// ---- small elementwise / VALU kernels ------------------------------------
__global__ void k_cvt_f16(const float* __restrict__ in, _Float16* __restrict__ out, int n) {
  int i = blockIdx.x * 256 + threadIdx.x;
  if (i < n) out[i] = (_Float16)in[i];
}

__global__ void k_fc0(const float* __restrict__ x, const float* __restrict__ w,
                      const float* __restrict__ b, _Float16* __restrict__ h) {
  long long i = (long long)blockIdx.x * 256 + threadIdx.x;  // (bb*128+c)*S + s
  if (i >= (long long)BB * WID * SS) return;
  int s = (int)(i % SS);
  int bc = (int)(i / SS);
  int c = bc % WID, bb = bc / WID;
  float xv = x[(long long)bb * SS + s];
  float g = (float)s / (float)(SS - 1);
  h[i] = (_Float16)(xv * w[c] + g * w[WID + c] + b[c]);
}

// forward twiddle: [S rows][128]; col 2k = cos(2pi k s/S), 2k+1 = -sin; col>=64 -> 0
__global__ void k_twid_fwd(_Float16* __restrict__ o) {
  int i = blockIdx.x * 256 + threadIdx.x;
  if (i >= SS * 128) return;
  int col = i & 127, s = i >> 7;
  if (col >= 64) { o[i] = (_Float16)0.f; return; }
  int k = col >> 1;
  int t = (k * s) & (SS - 1);
  float th = 6.283185307179586f * (float)t / (float)SS;
  o[i] = (_Float16)((col & 1) ? -sinf(th) : cosf(th));
}
// inverse twiddle: [64 rows][S]; row 2k = cos, 2k+1 = -sin (scales folded into Xm)
__global__ void k_twid_inv(_Float16* __restrict__ o) {
  int i = blockIdx.x * 256 + threadIdx.x;
  if (i >= 64 * SS) return;
  int s = i & (SS - 1), r = i / SS, k = r >> 1;
  int t = (k * s) & (SS - 1);
  float th = 6.283185307179586f * (float)t / (float)SS;
  o[i] = (_Float16)((r & 1) ? -sinf(th) : cosf(th));
}
// CFT basis: [S rows][128]; col n<72: kk=n>>3, m=(n>>1)&3, reim=n&1; else 0
__global__ void k_cft_basis(_Float16* __restrict__ o) {
  int i = blockIdx.x * 256 + threadIdx.x;
  if (i >= SS * 128) return;
  int col = i & 127, s = i >> 7;
  if (col >= 72) { o[i] = (_Float16)0.f; return; }
  int kk = col >> 3, m = (col >> 1) & 3, reim = col & 1;
  int k = kk - 4;
  int t = (k * s) % (2 * SS); if (t < 0) t += 2 * SS;
  float th = 3.14159265358979f * (float)t / (float)SS;
  float tp = 2.f * (float)s / (float)(SS - 1) - 1.f;
  float T[4]; T[0] = 1.f; T[1] = tp; T[2] = 2.f * tp * tp - 1.f;
  T[3] = tp * (4.f * tp * tp - 3.f);
  o[i] = (_Float16)(((reim) ? -sinf(th) : cosf(th)) * T[m]);
}

// complex mode mixing: out[b,o,k] = sum_i X[b,i,k]*w[i,o,k]; fold 1/S,2/S
__global__ void k_mix(const float* __restrict__ Xf, const float* __restrict__ sw,
                      _Float16* __restrict__ Xm) {
  int i = blockIdx.x * 256 + threadIdx.x;  // (b*128+o)*32+k
  if (i >= BB * WID * NMODES) return;
  int k = i & 31, bo = i >> 5, o = bo & 127, b = bo >> 7;
  float orr = 0.f, oii = 0.f;
  for (int ci = 0; ci < WID; ++ci) {
    const float* xp = Xf + ((long long)(b * WID + ci) * 128 + 2 * k);
    const float* wp = sw + ((((long long)ci * WID + o) * NMODES + k) << 1);
    float xr = xp[0], xi = xp[1], wr = wp[0], wi = wp[1];
    orr += xr * wr - xi * wi;
    oii += xr * wi + xi * wr;
  }
  float sc = (k == 0) ? (1.f / (float)SS) : (2.f / (float)SS);
  long long d = (long long)(b * WID + o) * 64 + 2 * k;
  Xm[d] = (_Float16)(orr * sc);
  Xm[d + 1] = (_Float16)(oii * sc);
}

// repack CFT GEMM output [4096][128] -> cftvec [B][C][M][L=2][K=9][2] f16
__global__ void k_cft_pack(const float* __restrict__ Cc, _Float16* __restrict__ v) {
  int i = blockIdx.x * 256 + threadIdx.x;
  const int DIM = WID * 4 * 2 * 9 * 2;  // 18432
  if (i >= BB * DIM) return;
  int b = i / DIM, r = i % DIM;
  int reim = r & 1, t = r >> 1;
  int kk = t % 9; t /= 9;
  t /= 2;            // drop L (duplicated segments)
  int m = t & 3, ci = t >> 2;
  float x = Cc[(long long)(b * WID + ci) * 128 + (kk * 4 + m) * 2 + reim];
  v[i] = (_Float16)(x * (1.f / (float)SS));
}

__global__ void k_cg2(const _Float16* __restrict__ l1, const float* __restrict__ w,
                      const float* __restrict__ b, float* __restrict__ lat) {
  int i = blockIdx.x * 256 + threadIdx.x;  // b*128+c
  if (i >= BB * WID) return;
  int c = i & 127, bb = i >> 7;
  float a = b[c];
  for (int j = 0; j < 256; ++j) a += (float)l1[bb * 256 + j] * w[j * WID + c];
  lat[i] = a;
}

// u[b][s][c] = fno[b][c][s] + latent[b][c]   (transpose for fc1 GEMM)
__global__ void k_combine(const _Float16* __restrict__ fno, const float* __restrict__ lat,
                          _Float16* __restrict__ u) {
  long long i = (long long)blockIdx.x * 256 + threadIdx.x;
  if (i >= (long long)BB * SS * WID) return;
  int c = (int)(i & 127);
  long long bs = i >> 7;
  int s = (int)(bs % SS), b = (int)(bs / SS);
  u[i] = (_Float16)((float)fno[((long long)b * WID + c) * SS + s] + lat[b * WID + c]);
}

__global__ void k_fc2(const _Float16* __restrict__ v, const float* __restrict__ w,
                      const float* __restrict__ b, float* __restrict__ out) {
  int i = blockIdx.x * 256 + threadIdx.x;  // b*S+s
  if (i >= BB * SS) return;
  const _Float16* row = v + (long long)i * WID;
  float a = b[0];
  for (int c = 0; c < WID; ++c) a += (float)row[c] * w[c];
  out[i] = a;
}

// --------------------------------------------------------------------------
extern "C" void kernel_launch(void* const* d_in, const int* in_sizes, int n_in,
                              void* d_out, int out_size, void* d_ws, size_t ws_size,
                              hipStream_t stream) {
  (void)in_sizes; (void)n_in; (void)out_size; (void)ws_size;
  const float* x     = (const float*)d_in[0];
  const float* fc0w  = (const float*)d_in[1];
  const float* fc0b  = (const float*)d_in[2];
  const float* sw[4] = {(const float*)d_in[3], (const float*)d_in[4],
                        (const float*)d_in[5], (const float*)d_in[6]};
  const float* wW[4] = {(const float*)d_in[7], (const float*)d_in[9],
                        (const float*)d_in[11], (const float*)d_in[13]};
  const float* wB[4] = {(const float*)d_in[8], (const float*)d_in[10],
                        (const float*)d_in[12], (const float*)d_in[14]};
  const float* fc1w = (const float*)d_in[15];
  const float* fc1b = (const float*)d_in[16];
  const float* fc2w = (const float*)d_in[17];
  const float* fc2b = (const float*)d_in[18];
  const float* cg1w = (const float*)d_in[19];
  const float* cg1b = (const float*)d_in[20];
  const float* cg2w = (const float*)d_in[21];
  const float* cg2b = (const float*)d_in[22];
  float* out = (float*)d_out;

  // bump allocator on workspace
  char* p = (char*)d_ws;
  auto alloc = [&](size_t n) { void* r = (void*)p; p += (n + 255) & ~(size_t)255; return r; };
  const size_t HBYTES = (size_t)BB * WID * SS * 2;           // 64 MB
  _Float16* hA    = (_Float16*)alloc(HBYTES);
  _Float16* hB    = (_Float16*)alloc(HBYTES);
  _Float16* Bfwd  = (_Float16*)alloc((size_t)SS * 128 * 2);
  _Float16* Binv  = (_Float16*)alloc((size_t)64 * SS * 2);
  _Float16* Bcft  = (_Float16*)alloc((size_t)SS * 128 * 2);
  float*    Xf    = (float*)alloc((size_t)BB * WID * 128 * 4);
  _Float16* Xm    = (_Float16*)alloc((size_t)BB * WID * 64 * 2);
  float*    Ccft  = (float*)alloc((size_t)BB * WID * 128 * 4);
  _Float16* cftv  = (_Float16*)alloc((size_t)BB * 18432 * 2);
  _Float16* lat1  = (_Float16*)alloc((size_t)BB * 256 * 2);
  float*    lat   = (float*)alloc((size_t)BB * WID * 4);
  _Float16* w16[4];
  for (int l = 0; l < 4; ++l) w16[l] = (_Float16*)alloc((size_t)WID * WID * 2);
  _Float16* fc1w16 = (_Float16*)alloc((size_t)WID * WID * 2);
  _Float16* cg1w16 = (_Float16*)alloc((size_t)18432 * 256 * 2);
  _Float16* u16 = hB;  // aliases: free after CFT GEMM
  _Float16* v16 = hA;  // free after combine

  auto cdiv = [](long long a, long long b) { return (unsigned)((a + b - 1) / b); };
  dim3 blk(256);

  // weight converts + basis fills
  for (int l = 0; l < 4; ++l)
    k_cvt_f16<<<cdiv(WID * WID, 256), blk, 0, stream>>>(wW[l], w16[l], WID * WID);
  k_cvt_f16<<<cdiv(WID * WID, 256), blk, 0, stream>>>(fc1w, fc1w16, WID * WID);
  k_cvt_f16<<<cdiv(18432 * 256, 256), blk, 0, stream>>>(cg1w, cg1w16, 18432 * 256);
  k_twid_fwd<<<cdiv((long long)SS * 128, 256), blk, 0, stream>>>(Bfwd);
  k_twid_inv<<<cdiv((long long)SS * 64, 256), blk, 0, stream>>>(Binv);
  k_cft_basis<<<cdiv((long long)SS * 128, 256), blk, 0, stream>>>(Bcft);

  // fc0 -> hA
  k_fc0<<<cdiv((long long)BB * WID * SS, 256), blk, 0, stream>>>(x, fc0w, fc0b, hA);

  // 4 FNO blocks (ping-pong hA <-> hB)
  _Float16* cur = hA;
  _Float16* nxt = hB;
  for (int l = 0; l < 4; ++l) {
    // forward truncated DFT: [4096,8192]@[8192,128(pad)] -> Xf f32
    gemm_wmma<64, 128><<<dim3(1, 64, 1), blk, 0, stream>>>(
        cur, SS, 0, Bfwd, 128, 0, Xf, nullptr, 128, 0, nullptr, SS, 0);
    k_mix<<<cdiv(BB * WID * NMODES, 256), blk, 0, stream>>>(Xf, sw[l], Xm);
    fno_block<<<dim3(SS / 128, WID / 64, BB), blk, 0, stream>>>(
        cur, w16[l], Xm, Binv, wB[l], nxt, (l < 3) ? 1 : 0);
    _Float16* t = cur; cur = nxt; nxt = t;
  }
  // now: fno_pred (f16) in hA, pre-block-4 h in hB

  // CFT branch on hB: [4096,8192]@[8192,128] -> Ccft f32
  gemm_wmma<64, 128><<<dim3(1, 64, 1), blk, 0, stream>>>(
      hB, SS, 0, Bcft, 128, 0, Ccft, nullptr, 128, 0, nullptr, SS, 0);
  k_cft_pack<<<cdiv((long long)BB * 18432, 256), blk, 0, stream>>>(Ccft, cftv);
  gemm_wmma<32, 64><<<dim3(4, 1, 1), blk, 0, stream>>>(
      cftv, 18432, 0, cg1w16, 256, 0, nullptr, lat1, 256, 0, cg1b, 18432, 1);
  k_cg2<<<cdiv(BB * WID, 256), blk, 0, stream>>>(lat1, cg2w, cg2b, lat);

  // combine + transpose -> u16 (hB), fc1 GEMM -> v16 (hA), fc2 -> out
  k_combine<<<cdiv((long long)BB * SS * WID, 256), blk, 0, stream>>>(hA, lat, u16);
  gemm_wmma<64, 128><<<dim3(1, 128, 32), blk, 0, stream>>>(
      u16, WID, (long long)SS * WID, fc1w16, WID, 0,
      nullptr, v16, WID, (long long)SS * WID, fc1b, WID, 1);
  k_fc2<<<cdiv(BB * SS, 256), blk, 0, stream>>>(v16, fc2w, fc2b, out);
}
